// MHSABlock_84851373900045
// MI455X (gfx1250) — matrix operand order repaired and checked by reading, not compiled, not run on hardware
//
#include <hip/hip_runtime.h>

// ---------------------------------------------------------------------------
// MHSA block for MI455X (gfx1250): bf16 WMMA (v_wmma_f32_16x16x32_bf16),
// wave32, LDS-tiled GEMMs, intermediates in bf16 padded to 224 columns.
// Block tile 128x64, 8 waves (4M x 2N), 32x32 output per wave = 4 WMMA/K-step.
// Double-buffered LDS pipeline: one barrier per K-step, global loads for the
// next tile issue before the barrier and overlap the current tile's WMMAs.
// ---------------------------------------------------------------------------

typedef __bf16 bf16_t;
typedef __bf16 v16bf __attribute__((ext_vector_type(16)));
typedef __bf16 v8bf  __attribute__((ext_vector_type(8)));
typedef float  v8f   __attribute__((ext_vector_type(8)));
typedef float  v4f   __attribute__((ext_vector_type(4)));

static constexpr float kEps = 1e-5f;

// Problem dims
static constexpr int BATCH = 64;
static constexpr int CIN   = 2048;
static constexpr int NSP   = 196;      // H*W
static constexpr int NP    = 224;      // padded spatial (multiple of 32)

// Workspace layout (byte offsets, all 16B aligned)
static constexpr size_t OFF_W1    = 0;                                   // 512*2048 bf16
static constexpr size_t OFF_WQKV  = OFF_W1   + (size_t)512*2048*2;       // 1536*512 bf16
static constexpr size_t OFF_W3    = OFF_WQKV + (size_t)1536*512*2;       // 2048*512 bf16
static constexpr size_t OFF_POS   = OFF_W3   + (size_t)2048*512*2;       // 512*224 bf16
static constexpr size_t OFF_OUT1  = OFF_POS  + (size_t)512*NP*2;         // 64*512*224 bf16
static constexpr size_t OFF_QKV   = OFF_OUT1 + (size_t)BATCH*512*NP*2;   // 64*1536*224 bf16 ([k|q|v])
static constexpr size_t OFF_AQ    = OFF_QKV  + (size_t)BATCH*1536*NP*2;  // 64*256*1024 bf16
static constexpr size_t OFF_SCORE = OFF_AQ   + (size_t)BATCH*256*1024*2; // 64*224*224 f32
static constexpr size_t OFF_ATTNT = OFF_SCORE+ (size_t)BATCH*NP*NP*4;    // 64*224*224 bf16
static constexpr size_t OFF_OUT2  = OFF_ATTNT+ (size_t)BATCH*NP*NP*2;    // 64*512*224 bf16

enum { EPI_BN_RELU = 0, EPI_QKV = 1, EPI_SCORES = 2, EPI_FINAL = 3 };

// ---------------------------------------------------------------------------
// Generic bf16 WMMA GEMM: C[b] = A (MxK bf16) * B[b] (KxN), block tile 128x64.
// ---------------------------------------------------------------------------
template<int EPI, typename TB>
__global__ __launch_bounds__(256) void gemm_wmma_k(
    const bf16_t* __restrict__ A, const TB* __restrict__ Bm, void* __restrict__ Out,
    const float* __restrict__ g, const float* __restrict__ bb,
    const float* __restrict__ mn, const float* __restrict__ vr,
    const float* __restrict__ resid,
    int K, int lda, int ldb, int nLimit, int ldo, int Mreal, int nReal,
    long sA, long sB, long sO)
{
    __shared__ __align__(16) bf16_t As[2 * 128 * 40];  // [buf][m][k], skewed
    __shared__ __align__(16) bf16_t Bs[2 * 64 * 40];   // [buf][n][k] (transposed)

    const int tid = threadIdx.x;
    const int bz  = blockIdx.z;
    const int n0  = blockIdx.x * 64;
    const int m0  = blockIdx.y * 128;
    A  += (size_t)bz * sA;
    Bm += (size_t)bz * sB;

    const int lane  = tid & 31;
    const int wave  = tid >> 5;
    const int waveM = wave >> 1;      // 0..3 -> 32 rows each
    const int waveN = wave & 1;       // 0..1 -> 32 cols each
    const int half  = lane >> 4;      // 0/1
    const int ln    = lane & 15;

    const int arow = tid >> 1, acol = (tid & 1) * 16;   // A staging: 16 bf16/thread
    const int brow = tid >> 3, bcol = (tid & 7) * 8;    // B staging: 8 elems/thread

    const bool fullN = (n0 + 64) <= nLimit;             // uniform per block
    const int  nk    = K >> 5;                          // K/32 tiles

    // ---- global -> register tile staging ----
    auto gload = [&](int k0, v8bf& a0r, v8bf& a1r, bf16_t (&br)[8]) {
        const bf16_t* srcA = A + (size_t)(m0 + arow) * lda + (k0 + acol);
        a0r = *(const v8bf*)srcA;
        a1r = *(const v8bf*)(srcA + 8);
        __builtin_prefetch(srcA + 32, 0, 3);            // near-scope prefetch
        const TB* srcRow = Bm + (size_t)(k0 + brow) * ldb;
        if (fullN) {
            const TB* src = srcRow + n0 + bcol;
            if constexpr (sizeof(TB) == 4) {            // fp32 source (x)
                v4f f0 = *(const v4f*)src;
                v4f f1 = *(const v4f*)(src + 4);
#pragma unroll
                for (int j = 0; j < 4; ++j) { br[j] = (bf16_t)f0[j]; br[4 + j] = (bf16_t)f1[j]; }
            } else {                                    // bf16 source
                v8bf b8 = *(const v8bf*)src;
#pragma unroll
                for (int j = 0; j < 8; ++j) br[j] = b8[j];
            }
        } else {                                        // clamped, branchless tail
#pragma unroll
            for (int j = 0; j < 8; ++j) {
                int nc  = n0 + bcol + j;
                int ncl = (nc < nLimit) ? nc : 0;
                float f = (float)srcRow[ncl];
                br[j]   = (bf16_t)((nc < nLimit) ? f : 0.0f);
            }
        }
    };
    // ---- register -> LDS tile store ----
    auto lstore = [&](int buf, v8bf a0r, v8bf a1r, const bf16_t (&br)[8]) {
        bf16_t* AsB = As + buf * (128 * 40);
        bf16_t* BsB = Bs + buf * (64 * 40);
        *(v8bf*)&AsB[arow * 40 + acol]     = a0r;
        *(v8bf*)&AsB[arow * 40 + acol + 8] = a1r;
#pragma unroll
        for (int j = 0; j < 8; ++j)
            BsB[(bcol + j) * 40 + brow] = br[j];
    };

    v8f acc[2][2] = {};
    v8bf a0r, a1r;
    bf16_t br[8];

    // prologue: tile 0 -> LDS buf 0
    gload(0, a0r, a1r, br);
    lstore(0, a0r, a1r, br);

    for (int it = 0; it < nk; ++it) {
        const int buf = it & 1;
        const bool hasNext = (it + 1) < nk;
        if (hasNext) gload((it + 1) * 32, a0r, a1r, br);   // overlap with compute
        __syncthreads();                                    // buf staged for all

        const bf16_t* AsB = As + buf * (128 * 40);
        const bf16_t* BsB = Bs + buf * (64 * 40);
        union Frag { v16bf v; v8bf h[2]; };
        Frag a[2], b[2];
#pragma unroll
        for (int sub = 0; sub < 2; ++sub) {
            const bf16_t* p = &AsB[(waveM * 32 + sub * 16 + ln) * 40 + 8 * half];
            a[sub].h[0] = *(const v8bf*)p;
            a[sub].h[1] = *(const v8bf*)(p + 16);
        }
#pragma unroll
        for (int jn = 0; jn < 2; ++jn) {
            const bf16_t* r = &BsB[(waveN * 32 + jn * 16 + ln) * 40 + 8 * half];
            b[jn].h[0] = *(const v8bf*)r;
            b[jn].h[1] = *(const v8bf*)(r + 16);
        }
#pragma unroll
        for (int jn = 0; jn < 2; ++jn)
#pragma unroll
            for (int sub = 0; sub < 2; ++sub)
                acc[sub][jn] = __builtin_amdgcn_wmma_f32_16x16x32_bf16(
                    false, a[sub].v, false, b[jn].v, (short)0, acc[sub][jn], false, false);

        if (hasNext) lstore((it + 1) & 1, a0r, a1r, br);   // fill other buffer
    }

    // ---- epilogue ----
#pragma unroll
    for (int sub = 0; sub < 2; ++sub) {
#pragma unroll
        for (int jn = 0; jn < 2; ++jn) {
            const int nn = n0 + waveN * 32 + jn * 16 + ln;
#pragma unroll
            for (int r = 0; r < 8; ++r) {
                int mr = m0 + waveM * 32 + sub * 16 + r + 8 * half;
                float val = acc[sub][jn][r];
                if constexpr (EPI == EPI_BN_RELU) {
                    if (nn < ldo) {
                        float s = g[mr] * rsqrtf(vr[mr] + kEps);
                        val = fmaxf(val * s + (bb[mr] - mn[mr] * s), 0.0f);
                        ((bf16_t*)Out)[(size_t)bz * sO + (size_t)mr * ldo + nn] =
                            (bf16_t)((nn < nReal) ? val : 0.0f);
                    }
                } else if constexpr (EPI == EPI_QKV) {
                    if (nn < ldo) {
                        int dr = (mr < 512) ? (512 + mr) : ((mr < 1024) ? (mr - 512) : mr);
                        ((bf16_t*)Out)[(size_t)bz * sO + (size_t)dr * ldo + nn] =
                            (bf16_t)((nn < nReal) ? val : 0.0f);
                    }
                } else if constexpr (EPI == EPI_SCORES) {
                    if (mr < Mreal && nn < ldo)
                        ((float*)Out)[(size_t)bz * sO + (size_t)mr * ldo + nn] = val;
                } else { // EPI_FINAL: bn3 + residual + relu, fp32, unpadded ld=196
                    if (nn < nReal) {
                        float s = g[mr] * rsqrtf(vr[mr] + kEps);
                        val = val * s + (bb[mr] - mn[mr] * s);
                        val += resid[(size_t)bz * sO + (size_t)mr * ldo + nn];
                        ((float*)Out)[(size_t)bz * sO + (size_t)mr * ldo + nn] =
                            fmaxf(val, 0.0f);
                    }
                }
            }
        }
    }
}

// ---------------------------------------------------------------------------
__global__ __launch_bounds__(256) void cvt_f32_bf16(const float* __restrict__ in,
                                                    bf16_t* __restrict__ out, int n) {
    int i = blockIdx.x * 256 + threadIdx.x;
    if (i < n) out[i] = (bf16_t)in[i];
}

__global__ __launch_bounds__(256) void pos_build(const float* __restrict__ rh,
                                                 const float* __restrict__ rw,
                                                 bf16_t* __restrict__ pos) {
    int i = blockIdx.x * 256 + threadIdx.x;   // 512*224
    int c = i / NP, n = i % NP;
    float v = 0.0f;
    if (n < NSP) { int h = n / 14, w = n % 14; v = rh[c * 14 + h] + rw[c * 14 + w]; }
    pos[i] = (bf16_t)v;
}

// Build A''[b][n][c] = (c<512 ? q[c][n] : pos[c-512][n]); rows padded to 256.
__global__ __launch_bounds__(256) void attn_prep(const bf16_t* __restrict__ qkv,
                                                 const bf16_t* __restrict__ pos,
                                                 bf16_t* __restrict__ Aq) {
    size_t i = (size_t)blockIdx.x * 256 + threadIdx.x;  // 64*224*1024
    int c = (int)(i & 1023);
    size_t r = i >> 10;
    int n = (int)(r % NP);
    int b = (int)(r / NP);
    bf16_t v = (c < 512)
        ? qkv[((size_t)b * 1536 + 512 + c) * NP + n]   // q stored at rows 512..1023
        : pos[(size_t)(c - 512) * NP + n];
    Aq[((size_t)b * 256 + n) * 1024 + c] = v;
}

// Row softmax of scores[b][n][0..195] -> attnT[b][m][n] (bf16), pad rows zeroed.
__global__ __launch_bounds__(256) void softmax_k(const float* __restrict__ S,
                                                 bf16_t* __restrict__ T) {
    int n = blockIdx.x, b = blockIdx.y, t = threadIdx.x;
    const float* row = S + ((size_t)b * NP + n) * NP;
    __shared__ float red[8];
    float x = (t < NSP) ? row[t] : -3.4e38f;
    for (int o = 16; o; o >>= 1) x = fmaxf(x, __shfl_xor(x, o, 32));
    if ((t & 31) == 0) red[t >> 5] = x;
    __syncthreads();
    float mx = red[0];
#pragma unroll
    for (int i = 1; i < 8; ++i) mx = fmaxf(mx, red[i]);
    __syncthreads();
    float e = (t < NSP) ? __expf(row[t] - mx) : 0.0f;
    float s = e;
    for (int o = 16; o; o >>= 1) s += __shfl_xor(s, o, 32);
    if ((t & 31) == 0) red[t >> 5] = s;
    __syncthreads();
    float tot = 0.0f;
#pragma unroll
    for (int i = 0; i < 8; ++i) tot += red[i];
    if (t < NP)
        T[((size_t)b * NP + t) * NP + n] = (t < NSP) ? (bf16_t)(e / tot) : (bf16_t)0.0f;
}

// ---------------------------------------------------------------------------
extern "C" void kernel_launch(void* const* d_in, const int* in_sizes, int n_in,
                              void* d_out, int out_size, void* d_ws, size_t ws_size,
                              hipStream_t stream) {
    const float* x     = (const float*)d_in[0];
    const float* w1    = (const float*)d_in[1];
    const float* g1    = (const float*)d_in[2];
    const float* b1    = (const float*)d_in[3];
    const float* m1    = (const float*)d_in[4];
    const float* v1    = (const float*)d_in[5];
    const float* wqkv  = (const float*)d_in[6];
    const float* rel_h = (const float*)d_in[7];
    const float* rel_w = (const float*)d_in[8];
    const float* g2    = (const float*)d_in[9];
    const float* b2    = (const float*)d_in[10];
    const float* m2    = (const float*)d_in[11];
    const float* v2    = (const float*)d_in[12];
    const float* w3    = (const float*)d_in[13];
    const float* g3    = (const float*)d_in[14];
    const float* b3    = (const float*)d_in[15];
    const float* m3    = (const float*)d_in[16];
    const float* v3    = (const float*)d_in[17];
    float* out = (float*)d_out;

    char* ws = (char*)d_ws;
    bf16_t* w1b   = (bf16_t*)(ws + OFF_W1);
    bf16_t* wqkvb = (bf16_t*)(ws + OFF_WQKV);
    bf16_t* w3b   = (bf16_t*)(ws + OFF_W3);
    bf16_t* posb  = (bf16_t*)(ws + OFF_POS);
    bf16_t* out1  = (bf16_t*)(ws + OFF_OUT1);
    bf16_t* qkvb  = (bf16_t*)(ws + OFF_QKV);
    bf16_t* Aq    = (bf16_t*)(ws + OFF_AQ);
    float*  score = (float*) (ws + OFF_SCORE);
    bf16_t* attnT = (bf16_t*)(ws + OFF_ATTNT);
    bf16_t* out2  = (bf16_t*)(ws + OFF_OUT2);

    // 0) one-time-per-call weight conversion + pos table
    cvt_f32_bf16<<<(512*2048 + 255)/256, 256, 0, stream>>>(w1, w1b, 512*2048);
    cvt_f32_bf16<<<(1536*512 + 255)/256, 256, 0, stream>>>(wqkv, wqkvb, 1536*512);
    cvt_f32_bf16<<<(2048*512 + 255)/256, 256, 0, stream>>>(w3, w3b, 2048*512);
    pos_build<<<(512*NP)/256, 256, 0, stream>>>(rel_h, rel_w, posb);

    // 1) out1 = relu(bn1(w1 @ x))          M=512 K=2048 (B reads fp32 x)
    gemm_wmma_k<EPI_BN_RELU, float><<<dim3(4, 4, BATCH), 256, 0, stream>>>(
        w1b, x, out1, g1, b1, m1, v1, nullptr,
        2048, 2048, NSP, NSP, NP, 512, NSP,
        0L, (long)CIN * NSP, (long)512 * NP);

    // 2) qkv = wqkv @ out1                 M=1536 K=512; rows reordered to [k|q|v]
    gemm_wmma_k<EPI_QKV, bf16_t><<<dim3(4, 12, BATCH), 256, 0, stream>>>(
        wqkvb, out1, qkvb, nullptr, nullptr, nullptr, nullptr, nullptr,
        512, 512, NP, NP, NP, 1536, NSP,
        0L, (long)512 * NP, (long)1536 * NP);

    // 3) A'' = [q^T | pos^T]  (196x1024 per batch, padded to 256 rows)
    attn_prep<<<(unsigned)((size_t)BATCH * NP * 1024 / 256), 256, 0, stream>>>(
        qkvb, posb, Aq);

    // 4) scores = A'' @ [k;q]              M=196(pad 256) K=1024, fuses cc+cp
    gemm_wmma_k<EPI_SCORES, bf16_t><<<dim3(4, 2, BATCH), 256, 0, stream>>>(
        Aq, qkvb, score, nullptr, nullptr, nullptr, nullptr, nullptr,
        1024, 1024, NP, NP, NP, NSP, NSP,
        (long)256 * 1024, (long)1536 * NP, (long)NP * NP);

    // 5) softmax rows -> attn^T (bf16, zero-padded K rows)
    softmax_k<<<dim3(NSP, BATCH), 256, 0, stream>>>(score, attnT);

    // 6) out2 = relu(bn2(v @ attn^T))      M=512 K=224
    gemm_wmma_k<EPI_BN_RELU, bf16_t><<<dim3(4, 4, BATCH), 256, 0, stream>>>(
        qkvb + (size_t)1024 * NP, attnT, out2, g2, b2, m2, v2, nullptr,
        NP, NP, NP, NP, NP, 512, NSP,
        (long)1536 * NP, (long)NP * NP, (long)512 * NP);

    // 7) out = relu(bn3(w3 @ out2) + x)    M=2048 K=512, fp32 output ld=196
    gemm_wmma_k<EPI_FINAL, bf16_t><<<dim3(4, 16, BATCH), 256, 0, stream>>>(
        w3b, out2, out, g3, b3, m3, v3, x,
        512, 512, NP, NP, NSP, 2048, NSP,
        0L, (long)512 * NP, (long)CIN * NSP);
}
